// Attention_5463198400554
// MI455X (gfx1250) — matrix operand order repaired and checked by reading, not compiled, not run on hardware
//
#include <hip/hip_runtime.h>

#define DIM   2048
#define BATCH 8

typedef __bf16 bf16_t;
typedef __attribute__((ext_vector_type(16))) __bf16 v16bf;
typedef __attribute__((ext_vector_type(8)))  float  v8f;

union Frag { unsigned u[8]; v16bf v; };

// pack two f32 into a bf16 pair (round-half-up) with one v_perm_b32
static __device__ __forceinline__ unsigned pack2(float lo, float hi) {
  unsigned a = __builtin_bit_cast(unsigned, lo) + 0x8000u;
  unsigned b = __builtin_bit_cast(unsigned, hi) + 0x8000u;
  // result: byte0,1 = lo bytes 2,3 ; byte2,3 = hi bytes 2,3
  return __builtin_amdgcn_perm(b, a, 0x07060302u);
}

// ---------------------------------------------------------------------------
// Kernel 1: v[b,c] = mean over n of x[b,n,c]
// ---------------------------------------------------------------------------
__global__ void col_mean_kernel(const float* __restrict__ x,
                                float* __restrict__ ws_v) {
  const int c = blockIdx.x * blockDim.x + threadIdx.x;   // 0..2047
  const int b = blockIdx.y;
  const float* xb = x + (size_t)b * DIM * DIM;
  float s = 0.f;
  for (int n = 0; n < DIM; ++n) s += xb[(size_t)n * DIM + c];
  ws_v[b * DIM + c] = s * (1.0f / (float)DIM);
}

// ---------------------------------------------------------------------------
// Kernel 2: G = x[b] * x[b]^T (symmetric; stored transposed == itself), bf16.
// One wave per 64x64 block (4x4 register tiles): 16 WMMAs per k-step share
// 8 fragment conversions -> ~12 VALU and 2 float4 loads per WMMA.
// ---------------------------------------------------------------------------
__global__ __launch_bounds__(256)
void gram_kernel(const float* __restrict__ x, bf16_t* __restrict__ G, int b0) {
  const int wave = threadIdx.x >> 5;
  const int lane = threadIdx.x & 31;
  const int half = lane >> 4;
  const int l16  = lane & 15;
  const int tile = blockIdx.x * 8 + wave;        // 0 .. 1023 (32x32 blocks)
  const int ib = (tile >> 5) * 64;
  const int mb = (tile & 31) * 64;
  const float* xb = x + (size_t)(b0 + blockIdx.y) * DIM * DIM;
  bf16_t* Gb = G + (size_t)blockIdx.y * DIM * DIM;

  const float* arows[4];
  const float* brows[4];
#pragma unroll
  for (int s = 0; s < 4; ++s) {
    arows[s] = xb + (size_t)(ib + s * 16 + l16) * DIM;  // A: M = l16
    brows[s] = xb + (size_t)(mb + s * 16 + l16) * DIM;  // B: N = l16
  }

  v8f acc[16];
  for (int t = 0; t < 16; ++t)
    for (int r = 0; r < 8; ++r) acc[t][r] = 0.f;

  for (int kc = 0; kc < DIM / 32; ++kc) {
    const int k0 = kc * 32;
    Frag A[4], B[4];
#pragma unroll
    for (int s = 0; s < 4; ++s) {
      // A fragment (16x32 bf16 layout): K = half*16 + e -> one 16-float run
      const float4* p0 = reinterpret_cast<const float4*>(arows[s] + k0 + half * 16);
      const float4* p1 = reinterpret_cast<const float4*>(arows[s] + k0 + half * 16 + 8);
      float4 r0 = p0[0], r1 = p0[1], r2 = p1[0], r3 = p1[1];
      A[s].u[0] = pack2(r0.x, r0.y); A[s].u[1] = pack2(r0.z, r0.w);
      A[s].u[2] = pack2(r1.x, r1.y); A[s].u[3] = pack2(r1.z, r1.w);
      A[s].u[4] = pack2(r2.x, r2.y); A[s].u[5] = pack2(r2.z, r2.w);
      A[s].u[6] = pack2(r3.x, r3.y); A[s].u[7] = pack2(r3.z, r3.w);
    }
#pragma unroll
    for (int s = 0; s < 4; ++s) {
      // B fragment (32x16 bf16 layout): K = half*16 + e -> one 16-float run
      const float4* p = reinterpret_cast<const float4*>(brows[s] + k0 + half * 16);
      float4 s0 = p[0], s1 = p[1], s2 = p[2], s3 = p[3];
      B[s].u[0] = pack2(s0.x, s0.y); B[s].u[1] = pack2(s0.z, s0.w);
      B[s].u[2] = pack2(s1.x, s1.y); B[s].u[3] = pack2(s1.z, s1.w);
      B[s].u[4] = pack2(s2.x, s2.y); B[s].u[5] = pack2(s2.z, s2.w);
      B[s].u[6] = pack2(s3.x, s3.y); B[s].u[7] = pack2(s3.z, s3.w);
    }
#pragma unroll
    for (int si = 0; si < 4; ++si)
#pragma unroll
      for (int sj = 0; sj < 4; ++sj)
        acc[si * 4 + sj] = __builtin_amdgcn_wmma_f32_16x16x32_bf16(
            false, A[si].v, false, B[sj].v, (short)0, acc[si * 4 + sj], false, false);
  }

  // Store transposed (G symmetric): lane writes G[m..][i..] rows, each subtile
  // as 8 packed bf16 = one 16-byte store.
#pragma unroll
  for (int si = 0; si < 4; ++si)
#pragma unroll
    for (int sj = 0; sj < 4; ++sj) {
      v8f a = acc[si * 4 + sj];
      uint4 st;
      st.x = pack2(a[0], a[1]);
      st.y = pack2(a[2], a[3]);
      st.z = pack2(a[4], a[5]);
      st.w = pack2(a[6], a[7]);
      *reinterpret_cast<uint4*>(
          Gb + (size_t)(mb + sj * 16 + l16) * DIM + ib + si * 16 + half * 8) = st;
    }
}

// ---------------------------------------------------------------------------
// Kernel 3: m[b,n] = (1/16) * max_m [ Wq * G ]_{n,m}
// One workgroup per (batch, 16-row n-tile); wave w owns m-tiles w*16..w*16+15.
// B fragments read G rows (contiguous bf16, zero conversion).
// ---------------------------------------------------------------------------
__global__ __launch_bounds__(256)
void rowmax_kernel(const float* __restrict__ Wq, const bf16_t* __restrict__ G,
                   float* __restrict__ ws_m, int b0) {
  __shared__ float red[8][16];
  const int tid  = threadIdx.x;
  const int wave = tid >> 5;
  const int lane = tid & 31;
  const int half = lane >> 4;
  const int l16  = lane & 15;
  const int n0   = blockIdx.x * 16;
  const int b    = b0 + blockIdx.y;
  const bf16_t* Gb = G + (size_t)blockIdx.y * DIM * DIM;
  const float* arow = Wq + (size_t)(n0 + l16) * DIM;

  v8f acc[16];
  for (int t = 0; t < 16; ++t)
    for (int r = 0; r < 8; ++r) acc[t][r] = 0.f;

  for (int kc = 0; kc < DIM / 32; ++kc) {
    const int k0 = kc * 32;
    Frag a;   // shared by all 16 m-tiles this k-step
    {
      const float4* p0 = reinterpret_cast<const float4*>(arow + k0 + half * 16);
      const float4* p1 = reinterpret_cast<const float4*>(arow + k0 + half * 16 + 8);
      float4 r0 = p0[0], r1 = p0[1], r2 = p1[0], r3 = p1[1];
      a.u[0] = pack2(r0.x, r0.y); a.u[1] = pack2(r0.z, r0.w);
      a.u[2] = pack2(r1.x, r1.y); a.u[3] = pack2(r1.z, r1.w);
      a.u[4] = pack2(r2.x, r2.y); a.u[5] = pack2(r2.z, r2.w);
      a.u[6] = pack2(r3.x, r3.y); a.u[7] = pack2(r3.z, r3.w);
    }
#pragma unroll
    for (int t = 0; t < 16; ++t) {
      const int m = (wave * 16 + t) * 16 + l16;   // B: N = m
      // B[K=i][N=m] = G[i,m] = G[m,i]  -> contiguous row read, already bf16
      const uint4* gp = reinterpret_cast<const uint4*>(
          Gb + (size_t)m * DIM + k0 + half * 16);
      Frag bm;
      uint4 g0 = gp[0], g1 = gp[1];
      bm.u[0] = g0.x; bm.u[1] = g0.y; bm.u[2] = g0.z; bm.u[3] = g0.w;
      bm.u[4] = g1.x; bm.u[5] = g1.y; bm.u[6] = g1.z; bm.u[7] = g1.w;
      acc[t] = __builtin_amdgcn_wmma_f32_16x16x32_bf16(
          false, a.v, false, bm.v, (short)0, acc[t], false, false);
    }
  }

  // Row-max: over this wave's 16 m-tiles, then across N lanes, then waves
  float vmax[8];
#pragma unroll
  for (int r = 0; r < 8; ++r) {
    float v = -3.402823466e38f;
#pragma unroll
    for (int t = 0; t < 16; ++t) v = fmaxf(v, acc[t][r]);
#pragma unroll
    for (int d = 1; d < 16; d <<= 1)
      v = fmaxf(v, __shfl_xor(v, d, 32));
    vmax[r] = v;
  }
  if (l16 == 0) {
#pragma unroll
    for (int r = 0; r < 8; ++r) red[wave][(half << 3) + r] = vmax[r];
  }
  __syncthreads();
  if (tid < 16) {
    float mv = -3.402823466e38f;
#pragma unroll
    for (int w = 0; w < 8; ++w) mv = fmaxf(mv, red[w][tid]);
    ws_m[(size_t)b * DIM + n0 + tid] = mv * 0.0625f;   // scale = 256^-0.5
  }
}

// ---------------------------------------------------------------------------
// Kernel 4: out[b,i,j] = v[b,i] * m[b,j]   (float4 stores)
// ---------------------------------------------------------------------------
__global__ void outer_kernel(const float* __restrict__ ws_v,
                             const float* __restrict__ ws_m,
                             float* __restrict__ out) {
  size_t idx = (size_t)blockIdx.x * blockDim.x + threadIdx.x;  // float4 index
  size_t e = idx * 4;
  int j = (int)(e & (DIM - 1));
  int i = (int)((e >> 11) & (DIM - 1));
  int b = (int)(e >> 22);
  float vi = ws_v[b * DIM + i];
  const float* mrow = ws_m + b * DIM;
  float4 o;
  o.x = vi * mrow[j + 0];
  o.y = vi * mrow[j + 1];
  o.z = vi * mrow[j + 2];
  o.w = vi * mrow[j + 3];
  *reinterpret_cast<float4*>(out + e) = o;
}

// ---------------------------------------------------------------------------
extern "C" void kernel_launch(void* const* d_in, const int* in_sizes, int n_in,
                              void* d_out, int out_size, void* d_ws, size_t ws_size,
                              hipStream_t stream) {
  (void)in_sizes; (void)n_in; (void)out_size;
  const float* x  = (const float*)d_in[0];
  const float* Wq = (const float*)d_in[1];
  float* out  = (float*)d_out;
  float* ws_v = (float*)d_ws;                       // BATCH*DIM floats
  float* ws_m = ws_v + BATCH * DIM;                 // BATCH*DIM floats
  bf16_t* G   = (bf16_t*)((char*)d_ws + 128 * 1024);

  const size_t g_batch_bytes = (size_t)DIM * DIM * sizeof(bf16_t);   // 8 MB
  const size_t need_full = 128 * 1024 + (size_t)BATCH * g_batch_bytes;

  dim3 g1(DIM / 256, BATCH);
  col_mean_kernel<<<g1, 256, 0, stream>>>(x, ws_v);

  const unsigned gram_blocks = (DIM / 64) * (DIM / 64) / 8;   // 128
  if (ws_size >= need_full) {
    // all batches in flight; G for every batch resident in ws (64 MB, L2-sized)
    gram_kernel<<<dim3(gram_blocks, BATCH), 256, 0, stream>>>(x, G, 0);
    rowmax_kernel<<<dim3(DIM / 16, BATCH), 256, 0, stream>>>(Wq, G, ws_m, 0);
  } else {
    // fallback: reuse one 8 MB G slab per batch (stream-ordered, deterministic)
    for (int b = 0; b < BATCH; ++b) {
      gram_kernel<<<dim3(gram_blocks, 1), 256, 0, stream>>>(x, G, b);
      rowmax_kernel<<<dim3(DIM / 16, 1), 256, 0, stream>>>(Wq, G, ws_m, b);
    }
  }

  const size_t total4 = (size_t)BATCH * DIM * DIM / 4;
  outer_kernel<<<(unsigned)(total4 / 256), 256, 0, stream>>>(ws_v, ws_m, out);
}
